// RSSM_81947976008236
// MI455X (gfx1250) — matrix operand order repaired and checked by reading, not compiled, not run on hardware
//
#include <hip/hip_runtime.h>

// ---------------------------------------------------------------------------
// RSSM rollout for MI455X (gfx1250, wave32, WMMA, async global->LDS copies).
// Weights -> bf16 per call (L2-resident, ~35MB << 192MB L2). Per step:
// bf16 WMMA GEMMs (f32 accum, 128x128 tiles, double-buffered async staging,
// split-K for narrow GEMMs) + fused elementwise kernels that also reduce
// the K-slices and add biases.
// ---------------------------------------------------------------------------

#define B_    512
#define T_    64
#define STOCH 64
#define DET   2048
#define HID   2048
#define ACT   32
#define GIN   (STOCH + ACT)   // 96 (padded to XK for GEMM)
#define XK    128             // padded K for the input GEMM
#define G3    (3 * DET)       // 6144

#define BM  128
#define BN  128
#define BK  64
#define LDP 72                // padded LDS row (shorts): 144B, 16B-aligned

typedef __attribute__((ext_vector_type(16))) __bf16 v16bf;
typedef __attribute__((ext_vector_type(8)))  float  v8f;

union FragBF { uint4 u[2]; v16bf v; };

__device__ __forceinline__ unsigned short f2bf(float f) {
  unsigned int x = __float_as_uint(f);
  x += 0x7FFFu + ((x >> 16) & 1u);   // RNE
  return (unsigned short)(x >> 16);
}

// ---- CDNA5 async global->LDS copy (16B per lane, tracked by ASYNCcnt) ----
// Inline asm (ISA 10.7 / VGLOBAL op 98): vdst = VGPR holding LDS byte offset,
// vaddr = 64-bit global address. Generic pointers to LDS carry the LDS byte
// offset in their low 32 bits (flat aperture rule), so truncation is exact.
__device__ __forceinline__ void async_ld_b128(unsigned short* lds, const unsigned short* g) {
  unsigned           off = (unsigned)(uintptr_t)lds;
  unsigned long long ga  = (unsigned long long)(uintptr_t)g;
  asm volatile("global_load_async_to_lds_b128 %0, %1, off" :: "v"(off), "v"(ga) : "memory");
}

template <int N>
__device__ __forceinline__ void wait_asynccnt() {
#if __has_builtin(__builtin_amdgcn_s_wait_asynccnt)
  __builtin_amdgcn_s_wait_asynccnt(N);
#else
  asm volatile("s_wait_asynccnt %0" :: "i"(N) : "memory");
#endif
}

// ---------------------------------------------------------------------------
// Generic strided f32 -> bf16 pack/convert
// ---------------------------------------------------------------------------
__global__ void pack_cvt_bf16(const float* __restrict__ src, long long s_stride, long long s_off,
                              unsigned short* __restrict__ dst, long long d_stride, long long d_off,
                              int rows, int cols) {
  int idx = blockIdx.x * blockDim.x + threadIdx.x;
  if (idx >= rows * cols) return;
  int r = idx / cols;
  int c = idx - r * cols;
  dst[(long long)r * d_stride + d_off + c] = f2bf(src[(long long)r * s_stride + s_off + c]);
}

__global__ void fill_u16(unsigned short* __restrict__ p, int n, unsigned short v) {
  int idx = blockIdx.x * blockDim.x + threadIdx.x;
  if (idx < n) p[idx] = v;
}

// ---------------------------------------------------------------------------
// bf16 WMMA GEMM slice:  C[z][M,N] = A[M,K](bf16) * W[N,K](bf16)^T over
// K-range [z*kps, (z+1)*kps).  No bias (consumers add it / reduce slices).
// 256 threads = 8 waves; block tile 128x128; wave tile 64x32 (4x2 frags).
// Double-buffered LDS, async global->LDS staging.
// ---------------------------------------------------------------------------
__launch_bounds__(256)
__global__ void gemm_bf16_wmma(const unsigned short* __restrict__ A, int lda,
                               const unsigned short* __restrict__ W, int ldw,
                               float* __restrict__ C, int N, int kps) {
  __shared__ __align__(16) unsigned short As[2][BM][LDP];
  __shared__ __align__(16) unsigned short Ws[2][BN][LDP];

  const int tid  = threadIdx.x;
  const int lane = tid & 31;
  const int wave = tid >> 5;
  const int half = (lane >> 4) & 1;
  const int ln16 = lane & 15;
  const int wm   = wave >> 2;          // 0..1  -> M offset wm*64
  const int wn   = wave & 3;           // 0..3  -> N offset wn*32
  const int m_base = blockIdx.y * BM;
  const int n_base = blockIdx.x * BN;
  const int k0     = blockIdx.z * kps;
  const int nIter  = kps / BK;

  v8f acc[4][2] = {};

  auto stage = [&](int bb, int kc) {
    // A: 128 rows x 64 cols = 1024 b128 chunks; W likewise. 8 per thread.
#pragma unroll
    for (int j = 0; j < 4; ++j) {
      int c = tid + j * 256;
      int row = c >> 3, seg = c & 7;
      async_ld_b128(&As[bb][row][seg * 8],
                    A + (size_t)(m_base + row) * lda + k0 + kc + seg * 8);
    }
#pragma unroll
    for (int j = 0; j < 4; ++j) {
      int c = tid + j * 256;
      int row = c >> 3, seg = c & 7;
      async_ld_b128(&Ws[bb][row][seg * 8],
                    W + (size_t)(n_base + row) * ldw + k0 + kc + seg * 8);
    }
  };

  stage(0, 0);
  for (int it = 0; it < nIter; ++it) {
    const int cur = it & 1;
    if (it + 1 < nIter) {            // prefetch next chunk into other buffer
      stage(cur ^ 1, (it + 1) * BK);
      wait_asynccnt<8>();            // oldest 8 (current chunk) retired
    } else {
      wait_asynccnt<0>();
    }
    __syncthreads();

#pragma unroll
    for (int ks = 0; ks < 2; ++ks) {  // two 32-wide sub-chunks of BK=64
      FragBF a[4], b[2];
#pragma unroll
      for (int mf = 0; mf < 4; ++mf) {
        int m = wm * 64 + mf * 16 + ln16;
        a[mf].u[0] = *reinterpret_cast<const uint4*>(&As[cur][m][32 * ks + 8 * half]);
        a[mf].u[1] = *reinterpret_cast<const uint4*>(&As[cur][m][32 * ks + 16 + 8 * half]);
      }
#pragma unroll
      for (int nf = 0; nf < 2; ++nf) {
        int n = wn * 32 + nf * 16 + ln16;
        b[nf].u[0] = *reinterpret_cast<const uint4*>(&Ws[cur][n][32 * ks + 8 * half]);
        b[nf].u[1] = *reinterpret_cast<const uint4*>(&Ws[cur][n][32 * ks + 16 + 8 * half]);
      }
#pragma unroll
      for (int mf = 0; mf < 4; ++mf)
#pragma unroll
        for (int nf = 0; nf < 2; ++nf)
          acc[mf][nf] = __builtin_amdgcn_wmma_f32_16x16x32_bf16(
              false, a[mf].v, false, b[nf].v, (short)0, acc[mf][nf], false, false);
    }
    __syncthreads();
  }

  float* Cz = C + (size_t)blockIdx.z * (size_t)B_ * N;
#pragma unroll
  for (int mf = 0; mf < 4; ++mf)
#pragma unroll
    for (int nf = 0; nf < 2; ++nf) {
      int ng = n_base + wn * 32 + nf * 16 + ln16;
#pragma unroll
      for (int r = 0; r < 8; ++r) {
        int mg = m_base + wm * 64 + mf * 16 + r + 8 * half;
        Cz[(size_t)mg * N + ng] = acc[mf][nf][r];
      }
    }
}

// ---------------------------------------------------------------------------
// GRU gate fusion: gi (1 slice) + gh (2 K-slices) + biases -> h_new.
// ---------------------------------------------------------------------------
__global__ void gru_fuse(const float* __restrict__ gi, const float* __restrict__ gh,
                         const float* __restrict__ b_ih, const float* __restrict__ b_hh,
                         const float* __restrict__ hprev, long long hprev_stride,
                         float* __restrict__ hout, long long hout_stride,
                         unsigned short* __restrict__ h_bf) {
  const size_t SL = (size_t)B_ * G3;
  int idx = blockIdx.x * blockDim.x + threadIdx.x;   // B*DET threads
  int b = idx >> 11;
  int j = idx & (DET - 1);
  const float* gib = gi + (size_t)b * G3;
  const float* gh0 = gh + (size_t)b * G3;
  const float* gh1 = gh0 + SL;
  float ir  = gib[j]            + b_ih[j];
  float iz  = gib[DET + j]      + b_ih[DET + j];
  float in_ = gib[2 * DET + j]  + b_ih[2 * DET + j];
  float hr  = gh0[j]            + gh1[j]           + b_hh[j];
  float hz  = gh0[DET + j]      + gh1[DET + j]     + b_hh[DET + j];
  float hn  = gh0[2 * DET + j]  + gh1[2 * DET + j] + b_hh[2 * DET + j];
  float r = 1.0f / (1.0f + expf(-(ir + hr)));
  float z = 1.0f / (1.0f + expf(-(iz + hz)));
  float n = tanhf(in_ + r * hn);
  float hp = hprev[(long long)b * hprev_stride + j];
  float h = (1.0f - z) * n + z * hp;
  hout[(long long)b * hout_stride + j] = h;
  h_bf[(size_t)b * DET + j] = f2bf(h);
}

// ---------------------------------------------------------------------------
// Reduce 4 K-slices + bias, LayerNorm(2048) + ELU, block-per-row, out bf16.
// ---------------------------------------------------------------------------
__global__ void ln_elu(const float* __restrict__ lin1, const float* __restrict__ pb1,
                       const float* __restrict__ gamma, const float* __restrict__ beta,
                       unsigned short* __restrict__ U) {
  __shared__ float rs[256];
  __shared__ float rq[256];
  const size_t SL = (size_t)B_ * HID;
  int b = blockIdx.x;
  float v[8];
  float s = 0.0f, q = 0.0f;
#pragma unroll
  for (int i = 0; i < 8; ++i) {
    int j = threadIdx.x + i * 256;
    float x = pb1[j];
#pragma unroll
    for (int sl = 0; sl < 4; ++sl) x += lin1[(size_t)sl * SL + (size_t)b * HID + j];
    v[i] = x; s += x; q += x * x;
  }
  rs[threadIdx.x] = s; rq[threadIdx.x] = q;
  __syncthreads();
  for (int o = 128; o > 0; o >>= 1) {
    if (threadIdx.x < o) { rs[threadIdx.x] += rs[threadIdx.x + o]; rq[threadIdx.x] += rq[threadIdx.x + o]; }
    __syncthreads();
  }
  float mu   = rs[0] * (1.0f / HID);
  float var  = rq[0] * (1.0f / HID) - mu * mu;
  float rstd = rsqrtf(var + 1e-5f);
#pragma unroll
  for (int i = 0; i < 8; ++i) {
    int j = threadIdx.x + i * 256;
    float u = (v[i] - mu) * rstd * gamma[j] + beta[j];
    u = u > 0.0f ? u : expm1f(u);    // ELU(alpha=1)
    U[(size_t)b * HID + j] = f2bf(u);
  }
}

// ---------------------------------------------------------------------------
// Reduce 16 K-slices + bias, softplus reparameterized sample.
// ---------------------------------------------------------------------------
__global__ void sample_prior(const float* __restrict__ out2, const float* __restrict__ pb2,
                             const float* __restrict__ noise_t, long long noise_stride,
                             float* __restrict__ s_out_t, long long s_stride,
                             unsigned short* __restrict__ x_bf) {
  const size_t SL = (size_t)B_ * 2 * STOCH;
  int idx = blockIdx.x * blockDim.x + threadIdx.x;   // B*STOCH threads
  int b = idx >> 6;
  int i = idx & (STOCH - 1);
  float mean = pb2[i];
  float ls   = pb2[STOCH + i];
#pragma unroll
  for (int ks = 0; ks < 16; ++ks) {
    mean += out2[(size_t)ks * SL + (size_t)b * 2 * STOCH + i];
    ls   += out2[(size_t)ks * SL + (size_t)b * 2 * STOCH + STOCH + i];
  }
  float sp  = ls > 20.0f ? ls : log1pf(expf(ls));    // stable softplus
  float std = sp + 0.1f;
  float eps = noise_t[(long long)b * noise_stride + i];
  float sn  = mean + eps * std;
  s_out_t[(long long)b * s_stride + i] = sn;
  x_bf[(size_t)b * XK + i] = f2bf(sn);
}

// ---------------------------------------------------------------------------
extern "C" void kernel_launch(void* const* d_in, const int* in_sizes, int n_in,
                              void* d_out, int out_size, void* d_ws, size_t ws_size,
                              hipStream_t stream) {
  const float* actions    = (const float*)d_in[0];
  const float* init_stoch = (const float*)d_in[1];
  const float* init_det   = (const float*)d_in[2];
  const float* noise      = (const float*)d_in[3];
  const float* W_ih       = (const float*)d_in[4];
  const float* b_ih       = (const float*)d_in[5];
  const float* W_hh       = (const float*)d_in[6];
  const float* b_hh       = (const float*)d_in[7];
  const float* pW1        = (const float*)d_in[8];
  const float* pb1        = (const float*)d_in[9];
  const float* p_gamma    = (const float*)d_in[10];
  const float* p_beta     = (const float*)d_in[11];
  const float* pW2        = (const float*)d_in[12];
  const float* pb2        = (const float*)d_in[13];

  float* S_out = (float*)d_out;                               // [B,T,STOCH]
  float* H_out = S_out + (size_t)B_ * T_ * STOCH;             // [B,T,DET]

  char*  ws  = (char*)d_ws;
  size_t off = 0;
  auto alloc = [&](size_t bytes) -> void* {
    off = (off + 255) & ~(size_t)255;
    void* p = ws + off;
    off += bytes;
    return p;
  };

  unsigned short* Wih_bf = (unsigned short*)alloc((size_t)G3 * XK * 2);     // K padded to 128
  unsigned short* Whh_bf = (unsigned short*)alloc((size_t)G3 * DET * 2);
  unsigned short* pW1_bf = (unsigned short*)alloc((size_t)HID * DET * 2);
  unsigned short* pW2_bf = (unsigned short*)alloc((size_t)(2 * STOCH) * HID * 2);
  unsigned short* x_bf   = (unsigned short*)alloc((size_t)B_ * XK * 2);     // K padded to 128
  unsigned short* h_bf   = (unsigned short*)alloc((size_t)B_ * DET * 2);
  unsigned short* u_bf   = (unsigned short*)alloc((size_t)B_ * HID * 2);
  float* gi   = (float*)alloc((size_t)B_ * G3 * 4);                          // 1 slice
  float* gh   = (float*)alloc((size_t)2 * B_ * G3 * 4);                      // 2 slices
  float* lin1 = (float*)alloc((size_t)4 * B_ * HID * 4);                     // 4 slices
  float* out2 = (float*)alloc((size_t)16 * B_ * 2 * STOCH * 4);              // 16 slices

  // Zero K-pad regions once per call, then convert weights to bf16.
  {
    int n = G3 * XK;
    fill_u16<<<(n + 255) / 256, 256, 0, stream>>>(Wih_bf, n, 0);
    n = B_ * XK;
    fill_u16<<<(n + 255) / 256, 256, 0, stream>>>(x_bf, n, 0);
  }
  pack_cvt_bf16<<<(G3 * GIN + 255) / 256, 256, 0, stream>>>(
      W_ih, GIN, 0, Wih_bf, XK, 0, G3, GIN);
  pack_cvt_bf16<<<(G3 * DET + 255) / 256, 256, 0, stream>>>(
      W_hh, DET, 0, Whh_bf, DET, 0, G3, DET);
  pack_cvt_bf16<<<(HID * DET + 255) / 256, 256, 0, stream>>>(
      pW1, DET, 0, pW1_bf, DET, 0, HID, DET);
  pack_cvt_bf16<<<(2 * STOCH * HID + 255) / 256, 256, 0, stream>>>(
      pW2, HID, 0, pW2_bf, HID, 0, 2 * STOCH, HID);
  pack_cvt_bf16<<<(B_ * STOCH + 255) / 256, 256, 0, stream>>>(
      init_stoch, STOCH, 0, x_bf, XK, 0, B_, STOCH);
  pack_cvt_bf16<<<(B_ * DET + 255) / 256, 256, 0, stream>>>(
      init_det, DET, 0, h_bf, DET, 0, B_, DET);

  const dim3 blk256(256);
  for (int t = 0; t < T_; ++t) {
    // x[:,64:96] = bf16(actions[:, t, :])  (cols 96..127 stay zero)
    pack_cvt_bf16<<<(B_ * ACT + 255) / 256, 256, 0, stream>>>(
        actions, (long long)T_ * ACT, (long long)t * ACT,
        x_bf, XK, STOCH, B_, ACT);

    // gi = x @ W_ih^T            [512 x 6144], K=128(pad), 1 slice
    gemm_bf16_wmma<<<dim3(G3 / BN, B_ / BM, 1), blk256, 0, stream>>>(
        x_bf, XK, Wih_bf, XK, gi, G3, XK);
    // gh = h @ W_hh^T            [512 x 6144], K=2048, 2 slices
    gemm_bf16_wmma<<<dim3(G3 / BN, B_ / BM, 2), blk256, 0, stream>>>(
        h_bf, DET, Whh_bf, DET, gh, G3, 1024);

    // GRU gates -> h_new (fp32 into H slice, bf16 for next GEMMs)
    const float* hprev = (t == 0) ? init_det : (H_out + (size_t)(t - 1) * DET);
    long long hprev_stride = (t == 0) ? (long long)DET : (long long)T_ * DET;
    gru_fuse<<<(B_ * DET) / 256, 256, 0, stream>>>(
        gi, gh, b_ih, b_hh, hprev, hprev_stride,
        H_out + (size_t)t * DET, (long long)T_ * DET, h_bf);

    // lin1 = h_new @ pW1^T       [512 x 2048], K=2048, 4 slices
    gemm_bf16_wmma<<<dim3(HID / BN, B_ / BM, 4), blk256, 0, stream>>>(
        h_bf, DET, pW1_bf, DET, lin1, HID, 512);

    // LayerNorm + ELU -> u (bf16), reduces the 4 slices + pb1
    ln_elu<<<B_, 256, 0, stream>>>(lin1, pb1, p_gamma, p_beta, u_bf);

    // out = u @ pW2^T            [512 x 128], K=2048, 16 slices
    gemm_bf16_wmma<<<dim3((2 * STOCH) / BN, B_ / BM, 16), blk256, 0, stream>>>(
        u_bf, HID, pW2_bf, HID, out2, 2 * STOCH, 128);

    // sample: reduces 16 slices + pb2 -> S slice + x stoch columns
    sample_prior<<<(B_ * STOCH) / 256, 256, 0, stream>>>(
        out2, pb2, noise + (size_t)t * STOCH, (long long)T_ * STOCH,
        S_out + (size_t)t * STOCH, (long long)T_ * STOCH, x_bf);
  }
}
